// MultiHeadAttention_89893665505782
// MI455X (gfx1250) — matrix operand order repaired and checked by reading, not compiled, not run on hardware
//
#include <hip/hip_runtime.h>
#include <hip/hip_bf16.h>
#include <stdint.h>

// ---------------- constants (from reference) ----------------
#define E_DIM 768
#define H_NUM 12
#define DH    64
#define B_NUM 8
#define S_LEN 1024
#define M_TOT (B_NUM * S_LEN)   // 8192

// ---------------- vector types ----------------
typedef __attribute__((ext_vector_type(16))) __bf16          v16bf;
typedef __attribute__((ext_vector_type(16))) unsigned short  v16u;
typedef __attribute__((ext_vector_type(8)))  float           v8f;
typedef __attribute__((ext_vector_type(4)))  float           v4f;
typedef __attribute__((ext_vector_type(4)))  unsigned int    v4u;
typedef __attribute__((ext_vector_type(2)))  unsigned int    v2u;

// GCC-style vector to match the async builtin's canonical parameter type
typedef int v4i __attribute__((vector_size(16)));
typedef __attribute__((address_space(1))) v4i* g_v4i_ptr;
typedef __attribute__((address_space(3))) v4i* l_v4i_ptr;

union FragBF { v16u u; v16bf b; v4u q[2]; };
union Pack8  { v4u q; unsigned short s[8]; };

// ---------------- CDNA5 async global->LDS (guarded) ----------------
#if defined(__has_builtin)
#if __has_builtin(__builtin_amdgcn_global_load_async_to_lds_b128)
#define HAVE_ASYNC 1
#endif
#endif
#ifndef HAVE_ASYNC
#define HAVE_ASYNC 0
#endif

__device__ __forceinline__ void cp16_g2l(unsigned short* lds, const unsigned short* g) {
#if HAVE_ASYNC
    __builtin_amdgcn_global_load_async_to_lds_b128(
        (g_v4i_ptr)g, (l_v4i_ptr)lds, 0, 0);
#else
    *(v4u*)lds = *(const v4u*)g;
#endif
}

__device__ __forceinline__ void wait_async() {
#if HAVE_ASYNC
#if __has_builtin(__builtin_amdgcn_s_wait_asynccnt)
    __builtin_amdgcn_s_wait_asynccnt(0);
#else
    asm volatile("s_wait_asynccnt 0x0" ::: "memory");
#endif
#endif
}

__device__ __forceinline__ unsigned short f2bf(float x) {
    __bf16 h = (__bf16)x;                     // native cvt on gfx1250
    return __builtin_bit_cast(unsigned short, h);
}
__device__ __forceinline__ unsigned int pack2(float lo, float hi) {
    return (unsigned int)f2bf(lo) | ((unsigned int)f2bf(hi) << 16);
}

__device__ __forceinline__ v8f wmma_bf16(v16bf a, v16bf b, v8f c) {
    return __builtin_amdgcn_wmma_f32_16x16x32_bf16(
        false, a, false, b, (short)0, c, false, false);
}

// ---- 16-lane row reductions via DPP row_ror (no LDS round-trips) ----
template <int CTRL>
__device__ __forceinline__ float dpp_ror_f32(float v) {
    return __builtin_bit_cast(float,
        __builtin_amdgcn_update_dpp(0, __builtin_bit_cast(int, v),
                                    CTRL, 0xf, 0xf, true));
}
__device__ __forceinline__ float red16_max(float v) {
    v = fmaxf(v, dpp_ror_f32<0x121>(v));   // row_ror:1
    v = fmaxf(v, dpp_ror_f32<0x122>(v));   // row_ror:2
    v = fmaxf(v, dpp_ror_f32<0x124>(v));   // row_ror:4
    v = fmaxf(v, dpp_ror_f32<0x128>(v));   // row_ror:8
    return v;
}
__device__ __forceinline__ float red16_sum(float v) {
    v += dpp_ror_f32<0x121>(v);
    v += dpp_ror_f32<0x122>(v);
    v += dpp_ror_f32<0x124>(v);
    v += dpp_ror_f32<0x128>(v);
    return v;
}

// A fragment from row-major [M][K] LDS tile (two contiguous 8-elem runs).
__device__ __forceinline__ v16bf afrag(const unsigned short* lds, int stride, int lane) {
    const int row = lane & 15, kg = lane >> 4;
    const unsigned short* p = lds + row * stride + kg * 8;
    FragBF f;
    f.q[0] = *(const v4u*)(p);
    f.q[1] = *(const v4u*)(p + 16);
    return f.b;
}

// B fragment from [N][K] LDS tile (two contiguous 8-elem runs at K = kg*8 and kg*8+16).
__device__ __forceinline__ v16bf bfragN(const unsigned short* lds, int stride, int lane) {
    const int n = lane & 15, kg = lane >> 4;
    const unsigned short* p = lds + n * stride + kg * 8;
    FragBF f;
    f.q[0] = *(const v4u*)(p);
    f.q[1] = *(const v4u*)(p + 16);
    return f.b;
}

// =====================================================================
// Kernel: flash attention.  grid = (B*H, S/128), block = 256.
// Async double-buffered K/V staging; Q A-frags hoisted; pipelined WMMA.
// =====================================================================
__global__ __launch_bounds__(256)
void attn_kernel(const unsigned short* __restrict__ Q,
                 const unsigned short* __restrict__ K,
                 const unsigned short* __restrict__ Vt,
                 unsigned short* __restrict__ Xattn) {
    const int bh   = blockIdx.x;
    const int qt   = blockIdx.y;
    const int b    = bh / H_NUM;
    const int h    = bh % H_NUM;
    const int q0   = qt * 128;
    const int tid  = threadIdx.x;
    const int wave = tid >> 5;
    const int lane = tid & 31;

    __shared__ __align__(16) unsigned short Qs[128][72];      // [M=s][K=d]
    __shared__ __align__(16) unsigned short Ks[2][64][72];    // [N=t][K=d]
    __shared__ __align__(16) unsigned short Vs[2][64][72];    // [N=d][K=t]
    __shared__ __align__(16) unsigned short Ps[8][16][72];    // wave scratch

    const size_t headBase = ((size_t)b * H_NUM + h) * S_LEN * DH;   // Q/K
    const size_t vBase    = ((size_t)b * H_NUM + h) * DH * S_LEN;   // Vt

    // prologue: stage Q tile and K/V tile 0 (async on CDNA5)
    for (int i = tid; i < 128 * 8; i += 256) {
        const int r = i >> 3, c4 = i & 7;
        cp16_g2l(&Qs[r][c4 * 8], Q + headBase + (size_t)(q0 + r) * DH + c4 * 8);
    }
    for (int i = tid; i < 64 * 8; i += 256) {
        const int r = i >> 3, c4 = i & 7;
        cp16_g2l(&Ks[0][r][c4 * 8], K + headBase + (size_t)r * DH + c4 * 8);
        cp16_g2l(&Vs[0][r][c4 * 8], Vt + vBase + (size_t)r * S_LEN + c4 * 8);
    }
    wait_async();
    __syncthreads();

    // hoist the Q A-fragments (invariant over the K/V loop)
    const v16bf a_q0 = afrag(&Qs[wave * 16][0],  72, lane);
    const v16bf a_q1 = afrag(&Qs[wave * 16][32], 72, lane);

    float m_i[8], l_i[8];
#pragma unroll
    for (int j = 0; j < 8; ++j) { m_i[j] = -1e30f; l_i[j] = 0.0f; }
    v8f o[4] = {};

    const int halfsel = (lane >> 4) << 3;
    const int ncol = lane & 15;

    int cur = 0;
    for (int t0 = 0; t0 < S_LEN; t0 += 64) {
        const int nxt = cur ^ 1;
        // kick off next tile's async copies while computing on current
        if (t0 + 64 < S_LEN) {
            for (int i = tid; i < 64 * 8; i += 256) {
                const int r = i >> 3, c4 = i & 7;
                cp16_g2l(&Ks[nxt][r][c4 * 8],
                         K + headBase + (size_t)(t0 + 64 + r) * DH + c4 * 8);
                cp16_g2l(&Vs[nxt][r][c4 * 8],
                         Vt + vBase + (size_t)r * S_LEN + t0 + 64 + c4 * 8);
            }
        }

        // S_ij = Q (16x64) x K^T : ping-pong pipelined fragment loads
        v8f sacc[4] = {};
        {
            v16bf bcur = bfragN(&Ks[cur][0][0], 72, lane);
#pragma unroll
            for (int i = 0; i < 8; ++i) {
                v16bf bnext = bcur;
                if (i < 7) {
                    const int i2 = i + 1;
                    bnext = bfragN(&Ks[cur][(i2 & 3) * 16][(i2 >> 2) * 32], 72, lane);
                }
                sacc[i & 3] = wmma_bf16((i >> 2) ? a_q1 : a_q0, bcur, sacc[i & 3]);
                bcur = bnext;
            }
        }

        // online softmax (row reductions in pure VALU via DPP)
#pragma unroll
        for (int j = 0; j < 8; ++j) {
            float mx = fmaxf(fmaxf(sacc[0][j], sacc[1][j]),
                             fmaxf(sacc[2][j], sacc[3][j]));
            mx = red16_max(mx);
            const float mnew = fmaxf(m_i[j], mx);
            const float fac  = __expf(m_i[j] - mnew);
            m_i[j] = mnew;
            float rs = 0.0f;
#pragma unroll
            for (int nt = 0; nt < 4; ++nt) {
                const float p = __expf(sacc[nt][j] - mnew);
                sacc[nt][j] = p;
                rs += p;
            }
            rs = red16_sum(rs);
            l_i[j] = l_i[j] * fac + rs;
#pragma unroll
            for (int nt = 0; nt < 4; ++nt)
                o[nt][j] *= fac;
        }

        // stage P into wave-private LDS slice (row-major [s][t])
#pragma unroll
        for (int nt = 0; nt < 4; ++nt)
#pragma unroll
            for (int j = 0; j < 8; ++j)
                Ps[wave][j + halfsel][nt * 16 + ncol] = f2bf(sacc[nt][j]);
        asm volatile("s_wait_dscnt 0" ::: "memory");

        // O += P (16x64) x V : ping-pong pipelined fragment loads
        {
            const v16bf a_p0 = afrag(&Ps[wave][0][0],  72, lane);
            const v16bf a_p1 = afrag(&Ps[wave][0][32], 72, lane);
            v16bf bcur = bfragN(&Vs[cur][0][0], 72, lane);
#pragma unroll
            for (int i = 0; i < 8; ++i) {
                v16bf bnext = bcur;
                if (i < 7) {
                    const int i2 = i + 1;
                    bnext = bfragN(&Vs[cur][(i2 & 3) * 16][(i2 >> 2) * 32], 72, lane);
                }
                o[i & 3] = wmma_bf16((i >> 2) ? a_p1 : a_p0, bcur, o[i & 3]);
                bcur = bnext;
            }
        }

        wait_async();        // next tile landed
        __syncthreads();     // everyone done with cur + staging visible
        cur = nxt;
    }

    // epilogue: normalize, restage through wave LDS, wide stores
#pragma unroll
    for (int j = 0; j < 8; ++j) {
        const float inv = 1.0f / l_i[j];
#pragma unroll
        for (int nt = 0; nt < 4; ++nt)
            Ps[wave][j + halfsel][nt * 16 + ncol] = f2bf(o[nt][j] * inv);
    }
    asm volatile("s_wait_dscnt 0" ::: "memory");
#pragma unroll
    for (int u = 0; u < 4; ++u) {
        const int idx = lane + u * 32;
        const int r16 = idx >> 3, c4 = idx & 7;
        const size_t row = (size_t)b * S_LEN + q0 + wave * 16 + r16;
        const v4u v = *(const v4u*)&Ps[wave][r16][c4 * 8];
        *(v4u*)&Xattn[row * E_DIM + h * DH + c4 * 8] = v;
    }
}

// =====================================================================
// Kernel: QKV projection.  grid = (M_TOT/128, H), block = 256.
// Q,K row-major [B,H,S,DH]; V TRANSPOSED [B,H,DH,S]. Q pre-scaled 1/8.
// =====================================================================
__global__ __launch_bounds__(256)
void qkv_kernel(const float* __restrict__ X,
                const float* __restrict__ Wq, const float* __restrict__ bq,
                const float* __restrict__ Wk, const float* __restrict__ bk,
                const float* __restrict__ Wv, const float* __restrict__ bv,
                unsigned short* __restrict__ Qo,
                unsigned short* __restrict__ Ko,
                unsigned short* __restrict__ Vt) {
    const int mt   = blockIdx.x;
    const int h    = blockIdx.y;
    const int m0   = mt * 128;
    const int tid  = threadIdx.x;
    const int wave = tid >> 5;
    const int lane = tid & 31;

    __shared__ __align__(16) unsigned short Xs[128][40];      // [M][K]
    __shared__ __align__(16) unsigned short Wst[3][64][40];   // [N=d][K=e]
    __shared__ __align__(16) unsigned short Os[8][16][72];    // wave scratch

    const float* Wm[3]   = { Wq, Wk, Wv };
    const float* bias[3] = { bq, bk, bv };

    v8f acc[3][4] = {};

    for (int kc = 0; kc < E_DIM; kc += 32) {
        // stage X (128x32 f32 -> bf16), 4-wide
        for (int i = tid; i < 128 * 8; i += 256) {
            const int r = i >> 3, c4 = i & 7;
            const v4f x = *(const v4f*)&X[(size_t)(m0 + r) * E_DIM + kc + c4 * 4];
            v2u p; p.x = pack2(x.x, x.y); p.y = pack2(x.z, x.w);
            *(v2u*)&Xs[r][c4 * 4] = p;
        }
        // stage W chunks transposed: Wst[m][d][e] (3*64*4 units, 3/thread)
        for (int unit = tid; unit < 768; unit += 256) {
            const int mm  = unit >> 8;
            const int rem = unit & 255;
            const int d   = rem & 63;
            const int e0  = (rem >> 6) * 8;
            const float* wp = Wm[mm] + (size_t)h * E_DIM * DH
                                     + (size_t)(kc + e0) * DH + d;
            Pack8 pk;
#pragma unroll
            for (int t = 0; t < 8; ++t)
                pk.s[t] = f2bf(wp[(size_t)t * DH]);
            *(v4u*)&Wst[mm][d][e0] = pk.q;
        }
        if (kc + 32 < E_DIM)
            __builtin_prefetch(&X[(size_t)(m0 + (tid >> 1)) * E_DIM + kc + 32 + (tid & 1) * 16], 0, 1);
        __syncthreads();

        // 12 WMMAs, ping-pong pipelined fragment loads
        const v16bf a = afrag(&Xs[wave * 16][0], 40, lane);
        v16bf bcur = bfragN(&Wst[0][0][0], 40, lane);
#pragma unroll
        for (int i = 0; i < 12; ++i) {
            v16bf bnext = bcur;
            if (i < 11) {
                const int i2 = i + 1;
                bnext = bfragN(&Wst[i2 >> 2][(i2 & 3) * 16][0], 40, lane);
            }
            acc[i >> 2][i & 3] = wmma_bf16(a, bcur, acc[i >> 2][i & 3]);
            bcur = bnext;
        }
        __syncthreads();
    }

    const int bI = m0 >> 10;
    const int s0 = m0 & 1023;
    const int halfsel = (lane >> 4) << 3;
    const int ncol = lane & 15;
    const size_t headRow = ((size_t)bI * H_NUM + h) * S_LEN + s0 + wave * 16;

    // Q and K: restage through wave-private LDS, then 16B global stores
    unsigned short* OmQK[2] = { Qo, Ko };
#pragma unroll
    for (int m = 0; m < 2; ++m) {
        const float scale = (m == 0) ? 0.125f : 1.0f;
#pragma unroll
        for (int nt = 0; nt < 4; ++nt) {
            const float bv_ = bias[m][h * DH + nt * 16 + ncol];
#pragma unroll
            for (int j = 0; j < 8; ++j)
                Os[wave][j + halfsel][nt * 16 + ncol] = f2bf((acc[m][nt][j] + bv_) * scale);
        }
        asm volatile("s_wait_dscnt 0" ::: "memory");
#pragma unroll
        for (int u = 0; u < 4; ++u) {
            const int idx = lane + u * 32;
            const int r16 = idx >> 3, c4 = idx & 7;
            const v4u v = *(const v4u*)&Os[wave][r16][c4 * 8];
            *(v4u*)&OmQK[m][(headRow + r16) * DH + c4 * 8] = v;
        }
    }
    // V: direct packed transposed stores -> Vt[b,h,d,s]
    {
        const int sbase = s0 + wave * 16 + halfsel;
#pragma unroll
        for (int nt = 0; nt < 4; ++nt) {
            const int d = nt * 16 + ncol;
            const float bv_ = bv[h * DH + d];
            Pack8 pk;
#pragma unroll
            for (int j = 0; j < 8; ++j)
                pk.s[j] = f2bf(acc[2][nt][j] + bv_);
            *(v4u*)&Vt[(((size_t)bI * H_NUM + h) * DH + d) * S_LEN + sbase] = pk.q;
        }
    }
}

// =====================================================================
// Kernel: output projection.  grid = (M_TOT/128, E/64), block = 256.
// =====================================================================
__global__ __launch_bounds__(256)
void proj_kernel(const unsigned short* __restrict__ Xa,
                 const float* __restrict__ Wo,
                 const float* __restrict__ bo,
                 float* __restrict__ out) {
    const int mt   = blockIdx.x;
    const int n0   = blockIdx.y * 64;
    const int m0   = mt * 128;
    const int tid  = threadIdx.x;
    const int wave = tid >> 5;
    const int lane = tid & 31;

    __shared__ __align__(16) unsigned short Xs[128][40];   // [M][K]
    __shared__ __align__(16) unsigned short Wst[64][40];   // [N][K]
    __shared__ __align__(16) float          Ofs[8][16][68];

    v8f acc[4] = {};

    for (int kc = 0; kc < E_DIM; kc += 32) {
        for (int i = tid; i < 128 * 4; i += 256) {
            const int r = i >> 2, c4 = i & 3;
            *(v4u*)&Xs[r][c4 * 8] =
                *(const v4u*)(Xa + (size_t)(m0 + r) * E_DIM + kc + c4 * 8);
        }
        {   // Wo transposed: 64 cols * 4 e-groups = 256 units, 1 per thread
            const int col = tid & 63;
            const int e0  = (tid >> 6) * 8;
            const float* wp = Wo + (size_t)(kc + e0) * E_DIM + n0 + col;
            Pack8 pk;
#pragma unroll
            for (int t = 0; t < 8; ++t)
                pk.s[t] = f2bf(wp[(size_t)t * E_DIM]);
            *(v4u*)&Wst[col][e0] = pk.q;
        }
        if (kc + 32 < E_DIM)
            __builtin_prefetch(Xa + (size_t)(m0 + (tid >> 1)) * E_DIM + kc + 32 + (tid & 1) * 16, 0, 1);
        __syncthreads();

        const v16bf a = afrag(&Xs[wave * 16][0], 40, lane);
        v16bf bcur = bfragN(&Wst[0][0], 40, lane);
#pragma unroll
        for (int i = 0; i < 4; ++i) {
            v16bf bnext = bcur;
            if (i < 3)
                bnext = bfragN(&Wst[(i + 1) * 16][0], 40, lane);
            acc[i] = wmma_bf16(a, bcur, acc[i]);
            bcur = bnext;
        }
        __syncthreads();
    }

    const int halfsel = (lane >> 4) << 3;
    const int ncol = lane & 15;
#pragma unroll
    for (int nt = 0; nt < 4; ++nt) {
        const float bias = bo[n0 + nt * 16 + ncol];
#pragma unroll
        for (int j = 0; j < 8; ++j)
            Ofs[wave][j + halfsel][nt * 16 + ncol] = acc[nt][j] + bias;
    }
    asm volatile("s_wait_dscnt 0" ::: "memory");
#pragma unroll
    for (int u = 0; u < 8; ++u) {
        const int idx = lane + u * 32;
        const int r16 = idx >> 4, c4 = idx & 15;
        const v4f v = *(const v4f*)&Ofs[wave][r16][c4 * 4];
        *(v4f*)&out[(size_t)(m0 + wave * 16 + r16) * E_DIM + n0 + c4 * 4] = v;
    }
}

// =====================================================================
extern "C" void kernel_launch(void* const* d_in, const int* in_sizes, int n_in,
                              void* d_out, int out_size, void* d_ws, size_t ws_size,
                              hipStream_t stream) {
    const float* hidden = (const float*)d_in[0];
    const float* Wq = (const float*)d_in[1];
    const float* bq = (const float*)d_in[2];
    const float* Wk = (const float*)d_in[3];
    const float* bk = (const float*)d_in[4];
    const float* Wv = (const float*)d_in[5];
    const float* bv = (const float*)d_in[6];
    const float* Wo = (const float*)d_in[7];
    const float* bo = (const float*)d_in[8];
    float* out = (float*)d_out;

    const size_t QKV_ELEMS = (size_t)B_NUM * H_NUM * S_LEN * DH;
    unsigned short* Qb  = (unsigned short*)d_ws;
    unsigned short* Kb  = Qb + QKV_ELEMS;
    unsigned short* Vtb = Kb + QKV_ELEMS;          // transposed [B,H,DH,S]
    unsigned short* Xb  = Vtb + QKV_ELEMS;
    (void)in_sizes; (void)n_in; (void)out_size; (void)ws_size;

    dim3 blk(256);

    dim3 g1(M_TOT / 128, H_NUM);
    qkv_kernel<<<g1, blk, 0, stream>>>(hidden, Wq, bq, Wk, bk, Wv, bv, Qb, Kb, Vtb);

    dim3 g2(B_NUM * H_NUM, S_LEN / 128);
    attn_kernel<<<g2, blk, 0, stream>>>(Qb, Kb, Vtb, Xb);

    dim3 g3(M_TOT / 128, E_DIM / 64);
    proj_kernel<<<g3, blk, 0, stream>>>(Xb, Wo, bo, out);
}
